// BiLSTM_73916387164505
// MI455X (gfx1250) — matrix operand order repaired and checked by reading, not compiled, not run on hardware
//
#include <hip/hip_runtime.h>
#include <hip/hip_bf16.h>

// ---------------------------------------------------------------------------
// BiLSTM(4x bidir, H=64) + CRF for MI455X (gfx1250, wave32, WMMA).
// Input projections = large bf16 WMMA GEMMs; recurrence = one workgroup per
// direction, LDS-resident state, branch-free gate math, vectorized (b128)
// gate traffic, next-step G prefetch overlapping the WMMA phase.
// ---------------------------------------------------------------------------

#define BB 32
#define TT 256
#define DD 768
#define HH 64
#define KK 9
#define MROWS (BB * TT) // 8192

typedef __attribute__((ext_vector_type(16))) __bf16 v16bf;
typedef __attribute__((ext_vector_type(8))) float v8f;
typedef __attribute__((ext_vector_type(16))) unsigned short v16u;
typedef __attribute__((ext_vector_type(8))) unsigned short v8u;

union ABf { v16bf v; v8u h[2]; };
union BBf { v16bf v; v16u u; };

__device__ __forceinline__ unsigned short f2bf(float x) {
  unsigned int u = __float_as_uint(x);
  unsigned int r = u + 0x7fffu + ((u >> 16) & 1u);
  return (unsigned short)(r >> 16);
}
__device__ __forceinline__ float sigf(float x) {
  return 1.0f / (1.0f + __expf(-x));
}
// branch-free tanh: no exec-mask divergence on the serial critical path
__device__ __forceinline__ float tanh_fast(float x) {
  float ax = __builtin_fabsf(x);
  float t  = __expf(-2.0f * ax);
  float r  = (1.0f - t) / (1.0f + t);
  return __builtin_copysignf(r, x);
}

// ------------------------- weight prep (f32 -> bf16) -----------------------
struct WP { const float* p[32]; };

__device__ __forceinline__ size_t wihOffD(int ld) {
  return (ld < 2) ? (size_t)ld * (256 * 768)
                  : (size_t)(2 * 256 * 768) + (size_t)(ld - 2) * (256 * 128);
}

__global__ void prep_kernel(WP wp, unsigned short* __restrict__ wih,
                            unsigned short* __restrict__ whh,
                            float* __restrict__ bias) {
  int ld = blockIdx.y;           // 0..7 = (layer, dir)
  int l  = ld >> 1;
  const float* Wih = wp.p[ld * 4 + 0];
  const float* Whh = wp.p[ld * 4 + 1];
  const float* bih = wp.p[ld * 4 + 2];
  const float* bhh = wp.p[ld * 4 + 3];
  int wihN = (l == 0 ? 768 : 128) * 256;
  size_t wo = wihOffD(ld);
  int stride = gridDim.x * blockDim.x;
  for (int i = blockIdx.x * blockDim.x + threadIdx.x; i < wihN; i += stride)
    wih[wo + i] = f2bf(Wih[i]);
  for (int i = blockIdx.x * blockDim.x + threadIdx.x; i < 256 * 64; i += stride)
    whh[(size_t)ld * (256 * 64) + i] = f2bf(Whh[i]);
  for (int i = blockIdx.x * blockDim.x + threadIdx.x; i < 256; i += stride)
    bias[ld * 256 + i] = bih[i] + bhh[i];
}

// ------------------------- embedding gather -> bf16 ------------------------
__global__ void embed_kernel(const int* __restrict__ ids,
                             const float* __restrict__ emb,
                             unsigned short* __restrict__ X0) {
  int row = blockIdx.x;                 // 0..8191 (b*T + t)
  size_t src = (size_t)ids[row] * DD;
  size_t dst = (size_t)row * DD;
  for (int j = threadIdx.x; j < DD; j += 256)
    X0[dst + j] = f2bf(emb[src + j]);
}

// ----------------- input projection GEMM: G = X * Wih^T + bias -------------
// M=8192, N=256, K=KD. Block = 8 waves; each wave owns a 16x64 output strip.
template <int KD>
__global__ __launch_bounds__(256) void gemm_kernel(
    const unsigned short* __restrict__ Xb,   // [M, KD] bf16
    const unsigned short* __restrict__ Wb,   // [256, KD] bf16 (row-major Wih)
    const float* __restrict__ bias,          // [256]
    float* __restrict__ G) {                 // [M, 256]
  const int lane = threadIdx.x & 31;
  const int w    = threadIdx.x >> 5;
  const int half = lane >> 4;
  const int lm   = lane & 15;
  const int m0   = blockIdx.x * 128 + w * 16;
  const int n0   = blockIdx.y * 64;

  v8f acc[4] = {};
  for (int kb = 0; kb < KD; kb += 32) {
    ABf a;
    const size_t arow = (size_t)(m0 + lm) * KD + kb + half * 16;
    a.h[0] = *(const v8u*)(Xb + arow);
    a.h[1] = *(const v8u*)(Xb + arow + 8);
    __builtin_prefetch(Xb + arow + 32, 0, 1);
#pragma unroll
    for (int j = 0; j < 4; ++j) {
      BBf bm;
      bm.u = *(const v16u*)(Wb + (size_t)(n0 + j * 16 + lm) * KD + kb + half * 16);
      acc[j] = __builtin_amdgcn_wmma_f32_16x16x32_bf16(
          false, a.v, false, bm.v, (short)0, acc[j], false, false);
    }
  }
#pragma unroll
  for (int j = 0; j < 4; ++j) {
    int n = n0 + j * 16 + lm;
    float bv = bias[n];
#pragma unroll
    for (int r = 0; r < 8; ++r) {
      int m = m0 + r + 8 * half;
      G[(size_t)m * 256 + n] = acc[j][r] + bv;
    }
  }
}

// ------------------------- recurrent LSTM layer ----------------------------
// grid = 2 (fw, bw). 8 waves. Per step: gates = G[t] + h*Whh^T (WMMA; Whh
// B-fragments are loop-invariant -> hoisted to VGPRs by the compiler), then
// fused branch-free gate nonlinearity with b128 gate traffic.
__global__ __launch_bounds__(256) void lstm_kernel(
    const float* __restrict__ Gfw, const float* __restrict__ Gbw,
    const unsigned short* __restrict__ WhhBase,  // bf16 [2][256][64]
    float* __restrict__ Hcat,                    // f32 [B,T,128]
    unsigned short* __restrict__ Xnext) {        // bf16 [B*T,128]
  __shared__ unsigned short hbf[32 * 64];
  __shared__ float cst[32 * 64];
  __shared__ float gt[32 * 256];

  const int dir = blockIdx.x;
  const float* Gd = dir ? Gbw : Gfw;
  const unsigned short* Whh = WhhBase + (size_t)dir * (256 * 64);
  const int tid  = threadIdx.x;
  const int lane = tid & 31;
  const int w    = tid >> 5;
  const int half = lane >> 4;
  const int lm   = lane & 15;
  const int mt   = w & 1;          // M-tile (batch rows 0-15 / 16-31)
  const int ntb  = (w >> 1) * 4;   // first of 4 N-tiles for this wave

  // elementwise ownership: 8 consecutive units per thread -> b128 traffic
  const int eb = tid >> 3;         // batch row 0..31
  const int eu = (tid & 7) * 8;    // unit base 0..56

  for (int i = tid; i < 32 * 64; i += 256) { hbf[i] = 0; cst[i] = 0.0f; }
  __syncthreads();

  for (int s = 0; s < TT; ++s) {
    int t = dir ? (TT - 1 - s) : s;

    // prefetch next step's gate rows so global loads overlap the WMMA phase
    if (s + 1 < TT) {
      int tn = dir ? (TT - 2 - s) : (s + 1);
      const float* gp = Gd + ((size_t)eb * TT + tn) * 256 + eu;
      __builtin_prefetch(gp, 0, 1);
      __builtin_prefetch(gp + 64, 0, 1);
      __builtin_prefetch(gp + 128, 0, 1);
      __builtin_prefetch(gp + 192, 0, 1);
    }

    v8f acc[4] = {};
#pragma unroll
    for (int kb = 0; kb < 64; kb += 32) {
      ABf a;
      int arow = (mt * 16 + lm) * 64 + kb + half * 16;
      a.h[0] = *(const v8u*)(hbf + arow);
      a.h[1] = *(const v8u*)(hbf + arow + 8);
#pragma unroll
      for (int j = 0; j < 4; ++j) {
        BBf bm;
        bm.u = *(const v16u*)(Whh + (size_t)((ntb + j) * 16 + lm) * 64 + kb + half * 16);
        acc[j] = __builtin_amdgcn_wmma_f32_16x16x32_bf16(
            false, a.v, false, bm.v, (short)0, acc[j], false, false);
      }
    }
#pragma unroll
    for (int j = 0; j < 4; ++j) {
      int n = (ntb + j) * 16 + lm;
#pragma unroll
      for (int r = 0; r < 8; ++r)
        gt[(mt * 16 + r + 8 * half) * 256 + n] = acc[j][r];
    }
    __syncthreads();

    {
      size_t grow = ((size_t)eb * TT + t) * 256 + eu;
      v8f gi = *(const v8f*)(gt + eb * 256 + eu)       + *(const v8f*)(Gd + grow);
      v8f gf = *(const v8f*)(gt + eb * 256 + 64 + eu)  + *(const v8f*)(Gd + grow + 64);
      v8f gg = *(const v8f*)(gt + eb * 256 + 128 + eu) + *(const v8f*)(Gd + grow + 128);
      v8f go = *(const v8f*)(gt + eb * 256 + 192 + eu) + *(const v8f*)(Gd + grow + 192);
      v8f cv = *(const v8f*)(cst + eb * 64 + eu);
      v8f hv;
      v8u hb;
#pragma unroll
      for (int e = 0; e < 8; ++e) {
        float c = sigf(gf[e]) * cv[e] + sigf(gi[e]) * tanh_fast(gg[e]);
        float h = sigf(go[e]) * tanh_fast(c);
        cv[e] = c;
        hv[e] = h;
        hb[e] = f2bf(h);
      }
      *(v8f*)(cst + eb * 64 + eu) = cv;
      *(v8u*)(hbf + eb * 64 + eu) = hb;
      size_t orow = ((size_t)eb * TT + t) * 128 + dir * 64 + eu;
      *(v8f*)(Hcat + orow) = hv;
      *(v8u*)(Xnext + orow) = hb;
    }
    __syncthreads();
  }
}

// ------------------------- compaction (valid_mask scatter) -----------------
__global__ void compact_kernel(const float* __restrict__ Hcat,
                               const int* __restrict__ valid,
                               const int* __restrict__ attn,
                               float* __restrict__ vout,
                               int* __restrict__ vmask) {
  __shared__ int tgt[TT];
  int b = blockIdx.x;
  if (threadIdx.x == 0) {
    int run = 0;
    for (int t = 0; t < TT; ++t) {
      int v = valid[b * TT + t];
      tgt[t] = v ? run : -1;
      run += v;
    }
  }
  __syncthreads();
  int t = threadIdx.x;
  int g = tgt[t];
  if (g >= 0) {
    const v8f* src = (const v8f*)(Hcat + ((size_t)b * TT + t) * 128);
    v8f* dst = (v8f*)(vout + ((size_t)b * TT + g) * 128);
#pragma unroll
    for (int j = 0; j < 16; ++j) dst[j] = src[j];
    vmask[b * TT + g] = (attn[b * TT + t] != 0) ? 1 : 0;
  }
}

// ------------------------- classifier logits -------------------------------
__global__ __launch_bounds__(256) void logits_kernel(
    const float* __restrict__ vout, const float* __restrict__ Wc,
    const float* __restrict__ bc, float* __restrict__ outBase) {
  __shared__ float w[KK * 128];
  __shared__ float bb[KK];
  for (int i = threadIdx.x; i < KK * 128; i += 256) w[i] = Wc[i];
  if (threadIdx.x < KK) bb[threadIdx.x] = bc[threadIdx.x];
  __syncthreads();
  int idx = blockIdx.x * 256 + threadIdx.x;   // 0..73727 exactly
  int row = idx / KK;
  int k   = idx - row * KK;
  const float* vr = vout + (size_t)row * 128;
  float acc = bb[k];
  for (int j = 0; j < 128; ++j) acc += vr[j] * w[k * 128 + j];
  outBase[1 + idx] = acc;
}

// ------------------------- CRF negative log-likelihood ---------------------
__global__ void crf_kernel(const float* __restrict__ lg,   // [B,T,K] logits
                           const int* __restrict__ labels,
                           const int* __restrict__ vmask,
                           const float* __restrict__ startT,
                           const float* __restrict__ endT,
                           const float* __restrict__ trans,
                           float* __restrict__ loss) {
  __shared__ float tr[KK * KK], st[KK], en[KK], part[BB];
  int tid = threadIdx.x;
  if (tid < KK * KK) tr[tid] = trans[tid];
  if (tid < KK) { st[tid] = startT[tid]; en[tid] = endT[tid]; }
  __syncthreads();
  int b = tid;  // 32 threads, one per batch row
  const int* mk = vmask + b * TT;
  const int* tg = labels + b * TT;

  // numerator (gold path score, torchcrf semantics)
  int t0 = tg[0]; if (t0 < 0) t0 = 0;
  float num = st[t0] + lg[((size_t)b * TT) * KK + t0];
  int seqlen = 0;
  for (int t = 0; t < TT; ++t) seqlen += (mk[t] != 0);
  for (int t = 1; t < TT; ++t) {
    if (mk[t]) {
      int tp = tg[t - 1]; if (tp < 0) tp = 0;
      int tc = tg[t];     if (tc < 0) tc = 0;
      num += tr[tp * KK + tc] + lg[((size_t)b * TT + t) * KK + tc];
    }
  }
  int lastT = tg[seqlen - 1]; if (lastT < 0) lastT = 0;
  num += en[lastT];

  // denominator (forward algorithm)
  float alpha[KK];
  for (int k = 0; k < KK; ++k)
    alpha[k] = st[k] + lg[((size_t)b * TT) * KK + k];
  for (int t = 1; t < TT; ++t) {
    if (!mk[t]) continue;
    float na[KK];
    for (int j = 0; j < KK; ++j) {
      float mx = -3.4e38f;
      for (int k = 0; k < KK; ++k) {
        float v = alpha[k] + tr[k * KK + j];
        mx = fmaxf(mx, v);
      }
      float ssum = 0.0f;
      for (int k = 0; k < KK; ++k)
        ssum += __expf(alpha[k] + tr[k * KK + j] - mx);
      na[j] = mx + __logf(ssum) + lg[((size_t)b * TT + t) * KK + j];
    }
    for (int j = 0; j < KK; ++j) alpha[j] = na[j];
  }
  float mx = -3.4e38f;
  for (int k = 0; k < KK; ++k) mx = fmaxf(mx, alpha[k] + en[k]);
  float ssum = 0.0f;
  for (int k = 0; k < KK; ++k) ssum += __expf(alpha[k] + en[k] - mx);
  float den = mx + __logf(ssum);

  part[b] = den - num;   // loss contribution = -(num - den)
  __syncthreads();
  if (b == 0) {
    float total = 0.0f;
    for (int i = 0; i < BB; ++i) total += part[i];
    loss[0] = total;
  }
}

// ---------------------------------------------------------------------------
static inline size_t wihOffH(int ld) {
  return (ld < 2) ? (size_t)ld * (256 * 768)
                  : (size_t)(2 * 256 * 768) + (size_t)(ld - 2) * (256 * 128);
}

extern "C" void kernel_launch(void* const* d_in, const int* in_sizes, int n_in,
                              void* d_out, int out_size, void* d_ws, size_t ws_size,
                              hipStream_t stream) {
  (void)in_sizes; (void)n_in; (void)out_size; (void)ws_size;
  const int*   ids    = (const int*)d_in[0];
  const int*   attn   = (const int*)d_in[1];
  const int*   valid  = (const int*)d_in[2];
  const int*   labels = (const int*)d_in[3];
  const float* emb    = (const float*)d_in[4];
  WP wp;
  for (int i = 0; i < 32; ++i) wp.p[i] = (const float*)d_in[5 + i];
  const float* Wcls = (const float*)d_in[37];
  const float* bcls = (const float*)d_in[38];
  const float* stT  = (const float*)d_in[39];
  const float* enT  = (const float*)d_in[40];
  const float* trn  = (const float*)d_in[41];
  float* outF = (float*)d_out;

  // workspace layout (bytes), 256-aligned chunks, total ~41.3 MB
  char* p = (char*)d_ws;
  unsigned short* X0    = (unsigned short*)(p + 0);          // 8192*768*2
  unsigned short* X12   = (unsigned short*)(p + 12582912);   // 8192*128*2
  float*          Gfw   = (float*)(p + 14680064);            // 8192*256*4
  float*          Gbw   = (float*)(p + 23068672);            // 8192*256*4
  unsigned short* Wihb  = (unsigned short*)(p + 31457280);   // 589824*2
  unsigned short* Whhb  = (unsigned short*)(p + 32636928);   // 131072*2
  float*          biasb = (float*)(p + 32899072);            // 2048*4
  float*          Hcat  = (float*)(p + 32907264);            // 8192*128*4
  float*          vout  = (float*)(p + 37101568);            // 8192*128*4
  int*            vmskW = (int*)(p + 41295872);              // 8192*4

  prep_kernel<<<dim3(96, 8), 256, 0, stream>>>(wp, Wihb, Whhb, biasb);
  embed_kernel<<<MROWS, 256, 0, stream>>>(ids, emb, X0);
  hipMemsetAsync(vout, 0, (size_t)MROWS * 128 * 4, stream);
  hipMemsetAsync(vmskW, 0, (size_t)MROWS * 4, stream);

  for (int l = 0; l < 4; ++l) {
    const unsigned short* Xin = (l == 0) ? X0 : X12;
    for (int dir = 0; dir < 2; ++dir) {
      const unsigned short* W = Wihb + wihOffH(l * 2 + dir);
      const float* bs = biasb + (l * 2 + dir) * 256;
      float* Gd = dir ? Gbw : Gfw;
      if (l == 0)
        gemm_kernel<768><<<dim3(64, 4), 256, 0, stream>>>(Xin, W, bs, Gd);
      else
        gemm_kernel<128><<<dim3(64, 4), 256, 0, stream>>>(Xin, W, bs, Gd);
    }
    lstm_kernel<<<2, 256, 0, stream>>>(Gfw, Gbw,
                                       Whhb + (size_t)l * 2 * (256 * 64),
                                       Hcat, X12);
  }

  compact_kernel<<<BB, 256, 0, stream>>>(Hcat, valid, attn, vout, vmskW);
  logits_kernel<<<288, 256, 0, stream>>>(vout, Wcls, bcls, outF);
  crf_kernel<<<1, 32, 0, stream>>>(outF + 1, labels, vmskW, stT, enT, trn, outF);
}
